// CustomSelfAttention_57140244906159
// MI455X (gfx1250) — matrix operand order repaired and checked by reading, not compiled, not run on hardware
//
#include <hip/hip_runtime.h>

#define EMBED 1024
#define NHEAD 16
#define HDIM  64
#define BB    2
#define TT    2048

typedef __attribute__((ext_vector_type(16))) __bf16 v16bf;
typedef __attribute__((ext_vector_type(8)))  __bf16 v8bf;
typedef __attribute__((ext_vector_type(8)))  float  v8f;
typedef __attribute__((ext_vector_type(4)))  float  f32x4;

static __device__ inline v8f wmma_bf16(v16bf a, v16bf b, v8f c) {
    return __builtin_amdgcn_wmma_f32_16x16x32_bf16(
        /*neg_a=*/false, a, /*neg_b=*/false, b,
        /*c_mod=*/(short)0, c, /*reuse_a=*/false, /*reuse_b=*/false);
}

static __device__ inline float redmax16(float x) {
    #pragma unroll
    for (int off = 1; off < 16; off <<= 1)
        x = fmaxf(x, __shfl_xor(x, off, 16));
    return x;
}
static __device__ inline float redsum16(float x) {
    #pragma unroll
    for (int off = 1; off < 16; off <<= 1)
        x += __shfl_xor(x, off, 16);
    return x;
}

// Async copy of one 16-byte chunk, global -> LDS (ASYNCcnt-tracked DMA).
static __device__ inline void async_copy_b128(const void* gaddr, unsigned lds_addr) {
    asm volatile("global_load_async_to_lds_b128 %0, %1, off"
                 :: "v"(lds_addr), "v"(gaddr) : "memory");
}
static __device__ inline void wait_async0() {
    asm volatile("s_wait_asynccnt 0x0" ::: "memory");
}

// ---------------------------------------------------------------------------
// Prep A: elementwise f32 -> bf16 (optionally scaled).  8 elems / thread.
// ---------------------------------------------------------------------------
__global__ __launch_bounds__(256) void pack_kernel(const float* __restrict__ src,
                                                   __bf16* __restrict__ dst,
                                                   float scale, int n8) {
    int i = blockIdx.x * 256 + threadIdx.x;
    if (i >= n8) return;
    const f32x4* p = (const f32x4*)(src + (size_t)i * 8);
    f32x4 a = p[0], b = p[1];
    v8bf o;
    #pragma unroll
    for (int j = 0; j < 4; j++) {
        o[j]     = (__bf16)(a[j] * scale);
        o[4 + j] = (__bf16)(b[j] * scale);
    }
    *(v8bf*)(dst + (size_t)i * 8) = o;
}

// ---------------------------------------------------------------------------
// Prep B: K [b,t,h*64+d] f32  ->  kT [b,h,d,t] bf16   (per-head transpose)
// ---------------------------------------------------------------------------
__global__ __launch_bounds__(256) void kt_kernel(const float* __restrict__ k,
                                                 __bf16* __restrict__ kT) {
    __shared__ float tile[32][33];
    const int tx = threadIdx.x, ty = threadIdx.y;
    const int tB = blockIdx.x * 32;      // time tile
    const int dB = blockIdx.y * 32;      // head-dim tile
    const int bh = blockIdx.z;
    const int b = bh / NHEAD, h = bh % NHEAD;
    #pragma unroll
    for (int r = 0; r < 4; r++) {
        int tL = ty + r * 8;
        tile[tL][tx] = k[(size_t)(b * TT + tB + tL) * EMBED + h * HDIM + dB + tx];
    }
    __syncthreads();
    #pragma unroll
    for (int r = 0; r < 4; r++) {
        int dL = ty + r * 8;
        kT[((size_t)bh * HDIM + dB + dL) * TT + tB + tx] = (__bf16)tile[tx][dL];
    }
}

// ---------------------------------------------------------------------------
// Prep C: Wo [out,in] f32 -> WoT [in,out] bf16
// ---------------------------------------------------------------------------
__global__ __launch_bounds__(256) void wot_kernel(const float* __restrict__ Wo,
                                                  __bf16* __restrict__ WoT) {
    __shared__ float tile[32][33];
    const int tx = threadIdx.x, ty = threadIdx.y;
    const int iB = blockIdx.x * 32;
    const int oB = blockIdx.y * 32;
    #pragma unroll
    for (int r = 0; r < 4; r++) {
        int oL = ty + r * 8;
        tile[oL][tx] = Wo[(size_t)(oB + oL) * EMBED + iB + tx];
    }
    __syncthreads();
    #pragma unroll
    for (int r = 0; r < 4; r++) {
        int iL = ty + r * 8;
        WoT[(size_t)(iB + iL) * EMBED + oB + tx] = (__bf16)tile[tx][iL];
    }
}

// ---------------------------------------------------------------------------
// Fused flash attention: bf16 inputs, async double-buffered K/V staging.
// Block = 4 waves = 64 query rows of one head; wave = 16 rows.
// ---------------------------------------------------------------------------
__global__ __launch_bounds__(128) void attn_kernel(const __bf16* __restrict__ qBF,
                                                   const __bf16* __restrict__ kT,
                                                   const __bf16* __restrict__ vBF,
                                                   __bf16* __restrict__ attnBF) {
    __shared__ __attribute__((aligned(32))) __bf16 sKt[2][64 * 64];  // [d][key]
    __shared__ __attribute__((aligned(32))) __bf16 sV [2][64 * 64];  // [key][d]
    __shared__ __attribute__((aligned(32))) __bf16 sP [4][16 * 64];  // per-wave P

    const int tid  = threadIdx.x;
    const int wave = tid >> 5;
    const int lane = tid & 31;
    const int bh   = blockIdx.y;
    const int b    = bh / NHEAD;
    const int h    = bh % NHEAD;
    const int qBase = blockIdx.x * 64 + wave * 16;
    const int m  = lane & 15;
    const int hi = lane >> 4;
    const int k0 = hi * 8;

    // ---- Q fragments (already bf16, pre-scaled by 1/8 in pack_kernel) ----
    const __bf16* qrow = qBF + (size_t)(b * TT + qBase + m) * EMBED + h * HDIM;
    v16bf qa[2];
    #pragma unroll
    for (int c = 0; c < 2; c++) {
        const v8bf lo = *(const v8bf*)(qrow + c * 32 + k0);
        const v8bf hv = *(const v8bf*)(qrow + c * 32 + k0 + 16);
        qa[c] = __builtin_shufflevector(lo, hv,
            0, 1, 2, 3, 4, 5, 6, 7, 8, 9, 10, 11, 12, 13, 14, 15);
    }

    float mrow[8], ssum[8];
    v8f acc[4];
    const v8f z8 = {0.f, 0.f, 0.f, 0.f, 0.f, 0.f, 0.f, 0.f};
    #pragma unroll
    for (int r = 0; r < 8; r++) { mrow[r] = -__builtin_inff(); ssum[r] = 0.f; }
    #pragma unroll
    for (int t = 0; t < 4; t++) acc[t] = z8;

    const __bf16* ktHead = kT  + (size_t)bh * HDIM * TT;              // [d][t]
    const __bf16* vHead  = vBF + (size_t)b * TT * EMBED + h * HDIM;   // [t][d]
    __bf16* myP = &sP[wave][0];

    // stage key-block kb into LDS buffer bufIdx: 1024 x 16B async chunks
    auto stage = [&](int kb, int bufIdx) {
        #pragma unroll
        for (int j = 0; j < 4; j++) {
            int c0 = j * 128 + tid;              // 0..511
            int row = c0 >> 3, piece = c0 & 7;   // 8 x 16B per 64-elem row
            // K^T tile row: d = row, 64 keys contiguous in global
            const __bf16* gk = ktHead + (size_t)row * TT + kb * 64 + piece * 8;
            async_copy_b128(gk, (unsigned)(uintptr_t)&sKt[bufIdx][row * 64 + piece * 8]);
            // V tile row: key = row, 64 dims contiguous in global
            const __bf16* gv = vHead + (size_t)(kb * 64 + row) * EMBED + piece * 8;
            async_copy_b128(gv, (unsigned)(uintptr_t)&sV[bufIdx][row * 64 + piece * 8]);
        }
    };

    stage(0, 0);

    for (int kb = 0; kb < TT / 64; kb++) {
        wait_async0();
        __syncthreads();
        if (kb + 1 < TT / 64) stage(kb + 1, (kb + 1) & 1);
        const int buf = kb & 1;

        // ---- S = Q * K^T ----
        v8f s[4];
        #pragma unroll
        for (int nb = 0; nb < 4; nb++) {
            v8f sc = z8;
            #pragma unroll
            for (int c = 0; c < 2; c++) {
                v16bf bk;
                #pragma unroll
                for (int s2 = 0; s2 < 16; s2++) {
                    int kk = c * 32 + (s2 < 8 ? k0 + s2 : 16 + k0 + s2 - 8);
                    bk[s2] = sKt[buf][kk * 64 + nb * 16 + m];
                }
                sc = wmma_bf16(qa[c], bk, sc);
            }
            s[nb] = sc;
        }

        // ---- online softmax (row = r + 8*hi, 16 cols across lane half) ----
        #pragma unroll
        for (int r = 0; r < 8; r++) {
            float tmax = fmaxf(fmaxf(s[0][r], s[1][r]), fmaxf(s[2][r], s[3][r]));
            tmax = redmax16(tmax);
            float mn = fmaxf(mrow[r], tmax);
            float alpha = __expf(mrow[r] - mn);
            mrow[r] = mn;
            ssum[r] = ssum[r] * alpha;
            #pragma unroll
            for (int t = 0; t < 4; t++) acc[t][r] = acc[t][r] * alpha;
            float psum = 0.f;
            #pragma unroll
            for (int nb = 0; nb < 4; nb++) {
                float pv = __expf(s[nb][r] - mn);
                psum += pv;
                myP[(r + 8 * hi) * 64 + nb * 16 + m] = (__bf16)pv;
            }
            ssum[r] += redsum16(psum);
        }

        // ---- acc += P * V ----
        #pragma unroll
        for (int c = 0; c < 2; c++) {
            const v8bf plo = *(const v8bf*)(myP + m * 64 + c * 32 + k0);
            const v8bf phi = *(const v8bf*)(myP + m * 64 + c * 32 + k0 + 16);
            v16bf pa = __builtin_shufflevector(plo, phi,
                0, 1, 2, 3, 4, 5, 6, 7, 8, 9, 10, 11, 12, 13, 14, 15);
            #pragma unroll
            for (int dt = 0; dt < 4; dt++) {
                v16bf bv;
                #pragma unroll
                for (int s2 = 0; s2 < 16; s2++) {
                    int kk = c * 32 + (s2 < 8 ? k0 + s2 : 16 + k0 + s2 - 8);
                    bv[s2] = sV[buf][kk * 64 + dt * 16 + m];
                }
                acc[dt] = wmma_bf16(pa, bv, acc[dt]);
            }
        }
    }

    // ---- normalize + store bf16 attn ----
    __bf16* orow = attnBF + (size_t)(b * TT + qBase) * EMBED + h * HDIM;
    #pragma unroll
    for (int r = 0; r < 8; r++) {
        float invs = 1.0f / ssum[r];
        int row = r + 8 * hi;
        #pragma unroll
        for (int dt = 0; dt < 4; dt++)
            orow[(size_t)row * EMBED + dt * 16 + m] = (__bf16)(acc[dt][r] * invs);
    }
}

// ---------------------------------------------------------------------------
// Projection: out = attn @ Wo^T + bo.  64x64 tile per block (4 waves).
// ---------------------------------------------------------------------------
__global__ __launch_bounds__(128) void proj_kernel(const __bf16* __restrict__ attnBF,
                                                   const __bf16* __restrict__ WoT,
                                                   const float* __restrict__ bo,
                                                   float* __restrict__ out) {
    const int tid  = threadIdx.x;
    const int wave = tid >> 5;
    const int lane = tid & 31;
    const int m  = lane & 15;
    const int hi = lane >> 4;
    const int k0 = hi * 8;
    const int rBase = blockIdx.y * 64 + wave * 16;
    const int oBase = blockIdx.x * 64;

    const v8f z8 = {0.f, 0.f, 0.f, 0.f, 0.f, 0.f, 0.f, 0.f};
    v8f acc[4];
    #pragma unroll
    for (int t = 0; t < 4; t++) acc[t] = z8;

    const __bf16* arow = attnBF + (size_t)(rBase + m) * EMBED;
    for (int ic = 0; ic < EMBED / 32; ic++) {
        const v8bf alo = *(const v8bf*)(arow + ic * 32 + k0);
        const v8bf ahi = *(const v8bf*)(arow + ic * 32 + k0 + 16);
        v16bf af = __builtin_shufflevector(alo, ahi,
            0, 1, 2, 3, 4, 5, 6, 7, 8, 9, 10, 11, 12, 13, 14, 15);
        const __bf16* brow = WoT + (size_t)(ic * 32 + lane) * EMBED + oBase;
        #pragma unroll
        for (int dt = 0; dt < 4; dt++) {
            v16bf bw;
            #pragma unroll
            for (int s2 = 0; s2 < 16; s2++) {
                int kk = ic * 32 + (s2 < 8 ? k0 + s2 : 16 + k0 + s2 - 8);
                bw[s2] = WoT[(size_t)kk * EMBED + oBase + dt * 16 + m];
            }
            acc[dt] = wmma_bf16(af, bw, acc[dt]);
        }
    }

    #pragma unroll
    for (int dt = 0; dt < 4; dt++) {
        float bias = bo[oBase + dt * 16 + m];
        #pragma unroll
        for (int r = 0; r < 8; r++) {
            int row = rBase + r + 8 * hi;
            out[(size_t)row * EMBED + oBase + dt * 16 + m] = acc[dt][r] + bias;
        }
    }
}

// ---------------------------------------------------------------------------
extern "C" void kernel_launch(void* const* d_in, const int* in_sizes, int n_in,
                              void* d_out, int out_size, void* d_ws, size_t ws_size,
                              hipStream_t stream) {
    (void)in_sizes; (void)n_in; (void)out_size; (void)ws_size;
    const float* q  = (const float*)d_in[0];
    const float* k  = (const float*)d_in[1];
    const float* v  = (const float*)d_in[2];
    const float* Wo = (const float*)d_in[3];
    const float* bo = (const float*)d_in[4];
    float* out = (float*)d_out;

    const size_t NE = (size_t)BB * TT * EMBED;   // 4.19M elems per tensor
    __bf16* wsb    = (__bf16*)d_ws;
    __bf16* qBF    = wsb;            // [B*T, E]  (pre-scaled)
    __bf16* kTp    = wsb + NE;       // [B,H,D,T]
    __bf16* vBFp   = wsb + 2 * NE;   // [B*T, E]
    __bf16* attnBF = wsb + 3 * NE;   // [B*T, E]
    __bf16* WoT    = wsb + 4 * NE;   // [E, E]

    const int n8 = (int)(NE / 8);
    pack_kernel<<<dim3((n8 + 255) / 256), 256, 0, stream>>>(q, qBF, 0.125f, n8);
    pack_kernel<<<dim3((n8 + 255) / 256), 256, 0, stream>>>(v, vBFp, 1.0f, n8);
    kt_kernel  <<<dim3(TT / 32, HDIM / 32, BB * NHEAD), dim3(32, 8), 0, stream>>>(k, kTp);
    wot_kernel <<<dim3(EMBED / 32, EMBED / 32), dim3(32, 8), 0, stream>>>(Wo, WoT);
    attn_kernel<<<dim3(TT / 64, BB * NHEAD), 128, 0, stream>>>(qBF, kTp, vBFp, attnBF);
    proj_kernel<<<dim3(EMBED / 64, (BB * TT) / 64), 128, 0, stream>>>(attnBF, WoT, bo, out);
}
